// MuAttention_48455821033856
// MI455X (gfx1250) — matrix operand order repaired and checked by reading, compiled, not run on hardware
//
#include <hip/hip_runtime.h>
#include <hip/hip_bf16.h>
#include <cstdint>
#include <cstddef>

// ---- problem constants (match reference) ----
#define NH    32
#define NKV   8
#define HD    128
#define BATCH 2
#define SEQ   2048
#define HID   4096
#define TOK   (BATCH * SEQ)

typedef __bf16 bf16;
typedef __attribute__((ext_vector_type(16))) __bf16 bf16x16;
typedef __attribute__((ext_vector_type(8)))  __bf16 bf16x8;
typedef __attribute__((ext_vector_type(4)))  __bf16 bf16x4;
typedef __attribute__((ext_vector_type(8)))  float  f32x8;
typedef __attribute__((ext_vector_type(4)))  float  f32x4;

// ---------------------------------------------------------------------------
// WMMA wrapper: D = A(16x32 bf16) * B(32x16 bf16) + C(16x16 f32)
// ---------------------------------------------------------------------------
static __device__ inline f32x8 wmma_bf16(bf16x16 a, bf16x16 b, f32x8 c) {
  return __builtin_amdgcn_wmma_f32_16x16x32_bf16(
      /*neg_a=*/false, a, /*neg_b=*/false, b,
      /*c_mod=*/(short)0, c, /*reuse_a=*/false, /*reuse_b=*/false);
}

// A-fragment (16x32, 16-bit): lane<16 -> row=lane, K={0..7,16..23};
// lane>=16 -> row=lane-16, K={8..15,24..31}.
static __device__ inline bf16x16 frag_a(const bf16* tile, int stride, int lane) {
  const bf16* p = tile + (size_t)(lane & 15) * stride + ((lane < 16) ? 0 : 8);
  bf16x8 lo = *(const bf16x8*)(p);
  bf16x8 hi = *(const bf16x8*)(p + 16);
  bf16x16 f;
#pragma unroll
  for (int i = 0; i < 8; ++i) { f[i] = lo[i]; f[i + 8] = hi[i]; }
  return f;
}

// B-fragment (32x16, 16-bit) from row-major [n][k] storage (column n of B is a
// contiguous K-run): lane<16 -> col=lane, K={0..7,16..23}; lane>=16 -> K={8..15,24..31}.
static __device__ inline bf16x16 frag_b(const bf16* tile, int stride, int lane) {
  const bf16* p = tile + (size_t)(lane & 15) * stride + ((lane < 16) ? 0 : 8);
  bf16x8 lo = *(const bf16x8*)(p);
  bf16x8 hi = *(const bf16x8*)(p + 16);
  bf16x16 f;
#pragma unroll
  for (int i = 0; i < 8; ++i) { f[i] = lo[i]; f[i + 8] = hi[i]; }
  return f;
}

// ---------------------------------------------------------------------------
// fp32 -> bf16 conversion (vector4)
// ---------------------------------------------------------------------------
__global__ void cvt_f32_bf16_v4(const float* __restrict__ in,
                                bf16* __restrict__ out, int n4) {
  int i = blockIdx.x * blockDim.x + threadIdx.x;
  int stride = gridDim.x * blockDim.x;
  for (; i < n4; i += stride) {
    f32x4 v = ((const f32x4*)in)[i];
    bf16x4 o;
    o[0] = (bf16)v[0]; o[1] = (bf16)v[1]; o[2] = (bf16)v[2]; o[3] = (bf16)v[3];
    ((bf16x4*)out)[i] = o;
  }
}

// ---------------------------------------------------------------------------
// V transpose:  Vb [B*S, NKV*HD]  ->  Vt [(b*NKV+kvh)*HD + d][SEQ]
// ---------------------------------------------------------------------------
__global__ void transpose_v(const bf16* __restrict__ Vb, bf16* __restrict__ Vt) {
  long i = (long)blockIdx.x * blockDim.x + threadIdx.x;  // flat over Vb
  const long total = (long)TOK * NKV * HD;
  if (i >= total) return;
  const int  d   = (int)(i & (HD - 1));
  const long r1  = i >> 7;
  const int  kvh = (int)(r1 & (NKV - 1));
  const long r2  = r1 >> 3;
  const int  s   = (int)(r2 & (SEQ - 1));
  const int  b   = (int)(r2 >> 11);
  Vt[(((size_t)b * NKV + kvh) * HD + d) * SEQ + s] = Vb[i];
}

// ---------------------------------------------------------------------------
// Tiled WMMA GEMM:  C[M,N] (f32 or bf16) = A[M,K] @ Bw[N,K]^T   (all bf16 in)
// Block tile 128(M)x256(N), 8 waves of 64x64 (4x4 accum tiles), K-step 32.
// ---------------------------------------------------------------------------
template <bool OUT_BF16>
__global__ __launch_bounds__(256) void gemm_wmma(
    const bf16* __restrict__ A, const bf16* __restrict__ Bw,
    void* __restrict__ Cvoid, int M, int N, int K) {
  __shared__ alignas(16) bf16 As[128][40];  // 32 used + 8 pad (16B-aligned rows)
  __shared__ alignas(16) bf16 Bs[256][40];

  const int lane = threadIdx.x & 31;
  const int wid  = threadIdx.x >> 5;
  const int wm   = wid & 1;   // 2 wave-rows of 64
  const int wn   = wid >> 1;  // 4 wave-cols of 64
  const int m0   = blockIdx.y * 128;
  const int n0   = blockIdx.x * 256;

  const int arow = threadIdx.x >> 1;        // 0..127 (2 threads per A row)
  const int acol = (threadIdx.x & 1) * 16;  // 0 or 16
  const int brow = threadIdx.x;             // 0..255 (1 thread per B row)

  f32x8 acc[4][4];
  const f32x8 z = {0.f, 0.f, 0.f, 0.f, 0.f, 0.f, 0.f, 0.f};
#pragma unroll
  for (int i = 0; i < 4; ++i)
#pragma unroll
    for (int j = 0; j < 4; ++j) acc[i][j] = z;

  const bf16* aRow = A  + (size_t)(m0 + arow) * K;
  const bf16* bRow = Bw + (size_t)(n0 + brow) * K;

  for (int k0 = 0; k0 < K; k0 += 32) {
    *(bf16x8*)&As[arow][acol]     = *(const bf16x8*)(aRow + k0 + acol);
    *(bf16x8*)&As[arow][acol + 8] = *(const bf16x8*)(aRow + k0 + acol + 8);
#pragma unroll
    for (int c = 0; c < 4; ++c)
      *(bf16x8*)&Bs[brow][c * 8] = *(const bf16x8*)(bRow + k0 + c * 8);

    if (k0 + 32 < K) {  // prefetch next K-slice (global_prefetch_b8)
      __builtin_prefetch(aRow + k0 + 32 + acol, 0, 1);
      __builtin_prefetch(bRow + k0 + 32, 0, 1);
    }
    __syncthreads();

    bf16x16 af[4], bfr[4];
#pragma unroll
    for (int i = 0; i < 4; ++i) af[i]  = frag_a(&As[wm * 64 + i * 16][0], 40, lane);
#pragma unroll
    for (int j = 0; j < 4; ++j) bfr[j] = frag_b(&Bs[wn * 64 + j * 16][0], 40, lane);

#pragma unroll
    for (int i = 0; i < 4; ++i)
#pragma unroll
      for (int j = 0; j < 4; ++j)
        acc[i][j] = wmma_bf16(af[i], bfr[j], acc[i][j]);
    __syncthreads();
  }

#pragma unroll
  for (int i = 0; i < 4; ++i)
#pragma unroll
    for (int j = 0; j < 4; ++j) {
      const int mBase = m0 + wm * 64 + i * 16;
      const int nIdx  = n0 + wn * 64 + j * 16 + (lane & 15);
#pragma unroll
      for (int r = 0; r < 8; ++r) {
        const int row = mBase + ((lane < 16) ? r : r + 8);
        if (OUT_BF16)
          ((bf16*)Cvoid)[(size_t)row * N + nIdx] = (bf16)acc[i][j][r];
        else
          ((float*)Cvoid)[(size_t)row * N + nIdx] = acc[i][j][r];
      }
    }
}

// ---------------------------------------------------------------------------
// Fused RMSNorm + RoPE, one wave per (token, head), hd=128 (4 elems/lane).
// ---------------------------------------------------------------------------
__global__ __launch_bounds__(256) void norm_rope(
    const float* __restrict__ X, const float* __restrict__ w,
    bf16* __restrict__ Y, int nheads) {
  const int lane = threadIdx.x & 31;
  const int wid  = threadIdx.x >> 5;
  const int idx  = blockIdx.x * 8 + wid;  // tok * nheads + head
  const int tok  = idx / nheads;
  const int head = idx - tok * nheads;
  const int pos  = tok & (SEQ - 1);

  const float* p = X + ((size_t)tok * nheads + head) * HD;
  float e0 = p[lane], e1 = p[lane + 32], e2 = p[lane + 64], e3 = p[lane + 96];

  float ss = e0 * e0 + e1 * e1 + e2 * e2 + e3 * e3;
#pragma unroll
  for (int m = 16; m >= 1; m >>= 1) ss += __shfl_xor(ss, m, 32);
  const float r = rsqrtf(ss * (1.0f / HD) + 1e-6f);

  e0 *= r * w[lane];      e1 *= r * w[lane + 32];
  e2 *= r * w[lane + 64]; e3 *= r * w[lane + 96];

  const float NLOG = -9.210340371976184f / 64.0f;  // -ln(10000)/64
  const float a0 = (float)pos * __expf((float)lane * NLOG);
  const float a1 = (float)pos * __expf((float)(lane + 32) * NLOG);
  float c0, s0, c1, s1;
  __sincosf(a0, &s0, &c0);
  __sincosf(a1, &s1, &c1);

  bf16* q = Y + ((size_t)tok * nheads + head) * HD;
  q[lane]      = (bf16)(e0 * c0 - e2 * s0);
  q[lane + 64] = (bf16)(e2 * c0 + e0 * s0);
  q[lane + 32] = (bf16)(e1 * c1 - e3 * s1);
  q[lane + 96] = (bf16)(e3 * c1 + e1 * s1);
}

// ---------------------------------------------------------------------------
// Flash attention (causal, GQA 4:1). grid = (SEQ/128, NH, BATCH), 256 thr.
// Each wave owns 16 q rows x hd=128. B-fragment loads are software-pipelined
// (2-deep) so each global b128 pair overlaps the preceding WMMA.
// ---------------------------------------------------------------------------
__global__ __launch_bounds__(256) void flash_attn(
    const bf16* __restrict__ Q, const bf16* __restrict__ K,
    const bf16* __restrict__ Vt, bf16* __restrict__ O) {
  __shared__ alignas(16) bf16 Plds[8][16][136];  // per-wave 16x128 (+8 pad)

  const int lane = threadIdx.x & 31;
  const int wid  = threadIdx.x >> 5;
  const int h    = blockIdx.y;
  const int b    = blockIdx.z;
  const int kvh  = h >> 2;
  const int q0   = blockIdx.x * 128 + wid * 16;
  const int ldq  = NH * HD;
  const int ldk  = NKV * HD;

  const bf16* qbase  = Q  + ((size_t)(b * SEQ + q0)) * ldq + h * HD;
  const bf16* vtbase = Vt + ((size_t)(b * NKV + kvh)) * HD * SEQ;  // [d][s]

  bf16x16 aq[4];
#pragma unroll
  for (int kc = 0; kc < 4; ++kc) aq[kc] = frag_a(qbase + kc * 32, ldq, lane);

  f32x8 ot[8];
  float mrun[8], lrun[8];
  const f32x8 z = {0.f, 0.f, 0.f, 0.f, 0.f, 0.f, 0.f, 0.f};
#pragma unroll
  for (int t = 0; t < 8; ++t) { ot[t] = z; mrun[t] = -3.0e38f; lrun[t] = 0.f; }

  const float scale = 0.08838834764831845f;  // 1/sqrt(128)

  for (int kb = 0; kb <= (int)blockIdx.x; ++kb) {
    const bf16* kbase = K + ((size_t)(b * SEQ + kb * 128)) * ldk + kvh * HD;

    // ---- scores: kc outer, nt inner; pipelined K-fragment loads ----
    f32x8 sc[8];
#pragma unroll
    for (int nt = 0; nt < 8; ++nt) sc[nt] = z;
#pragma unroll
    for (int kc = 0; kc < 4; ++kc) {
      bf16x16 bk[2];
      bk[0] = frag_b(kbase + kc * 32, ldk, lane);  // nt = 0
#pragma unroll
      for (int nt = 0; nt < 8; ++nt) {
        if (nt < 7)
          bk[(nt + 1) & 1] =
              frag_b(kbase + (size_t)((nt + 1) * 16) * ldk + kc * 32, ldk, lane);
        sc[nt] = wmma_bf16(aq[kc], bk[nt & 1], sc[nt]);
      }
    }

    // ---- scale + causal mask (diagonal block only) ----
#pragma unroll
    for (int nt = 0; nt < 8; ++nt)
#pragma unroll
      for (int r = 0; r < 8; ++r) sc[nt][r] *= scale;

    if (kb == (int)blockIdx.x) {
#pragma unroll
      for (int nt = 0; nt < 8; ++nt) {
        const int ki = kb * 128 + nt * 16 + (lane & 15);
#pragma unroll
        for (int r = 0; r < 8; ++r) {
          const int qi = q0 + ((lane < 16) ? r : r + 8);
          if (ki > qi) sc[nt][r] = -3.0e38f;
        }
      }
    }

    // ---- online softmax (row reductions across 16-lane groups) ----
#pragma unroll
    for (int r = 0; r < 8; ++r) {
      float mx = -3.0e38f;
#pragma unroll
      for (int nt = 0; nt < 8; ++nt) mx = fmaxf(mx, sc[nt][r]);
#pragma unroll
      for (int m = 8; m >= 1; m >>= 1) mx = fmaxf(mx, __shfl_xor(mx, m, 16));
      const float mnew = fmaxf(mrun[r], mx);
      const float corr = __expf(mrun[r] - mnew);
      mrun[r] = mnew;
      float rs = 0.f;
#pragma unroll
      for (int nt = 0; nt < 8; ++nt) {
        const float pv = __expf(sc[nt][r] - mnew);
        sc[nt][r] = pv;
        rs += pv;
      }
#pragma unroll
      for (int m = 8; m >= 1; m >>= 1) rs += __shfl_xor(rs, m, 16);
      lrun[r] = lrun[r] * corr + rs;
#pragma unroll
      for (int ht = 0; ht < 8; ++ht) ot[ht][r] *= corr;
    }

    // ---- stage P (bf16) into per-wave LDS (D-layout -> row-major) ----
#pragma unroll
    for (int nt = 0; nt < 8; ++nt) {
      const int kcol = nt * 16 + (lane & 15);
#pragma unroll
      for (int r = 0; r < 8; ++r) {
        const int m = (lane < 16) ? r : r + 8;
        Plds[wid][m][kcol] = (bf16)sc[nt][r];
      }
    }

    // ---- O += P @ V : pipelined Vt-fragment loads ----
#pragma unroll
    for (int kc = 0; kc < 4; ++kc) {
      bf16x16 ap = frag_a(&Plds[wid][0][kc * 32], 136, lane);
      const bf16* vcol = vtbase + (size_t)kb * 128 + kc * 32;
      bf16x16 bv[2];
      bv[0] = frag_b(vcol, SEQ, lane);  // ht = 0
#pragma unroll
      for (int ht = 0; ht < 8; ++ht) {
        if (ht < 7)
          bv[(ht + 1) & 1] = frag_b(vcol + (size_t)((ht + 1) * 16) * SEQ, SEQ, lane);
        ot[ht] = wmma_bf16(ap, bv[ht & 1], ot[ht]);
      }
    }
  }

  // ---- epilogue: normalize by l, store bf16 ----
  bf16* obase = O + ((size_t)(b * SEQ + q0)) * ldq + h * HD;
#pragma unroll
  for (int r = 0; r < 8; ++r) {
    const int m = (lane < 16) ? r : r + 8;
    const float inv = 1.0f / lrun[r];
#pragma unroll
    for (int ht = 0; ht < 8; ++ht)
      obase[(size_t)m * ldq + ht * 16 + (lane & 15)] = (bf16)(ot[ht][r] * inv);
  }
}

// ---------------------------------------------------------------------------
// Host launcher
// ---------------------------------------------------------------------------
extern "C" void kernel_launch(void* const* d_in, const int* in_sizes, int n_in,
                              void* d_out, int out_size, void* d_ws, size_t ws_size,
                              hipStream_t stream) {
  (void)in_sizes; (void)n_in; (void)out_size; (void)ws_size;

  const float* hs = (const float*)d_in[0];
  // d_in[1] = position_ids (arange(S) broadcast) -> pos = tok % SEQ, unused
  const float* Wq = (const float*)d_in[2];
  const float* Wk = (const float*)d_in[3];
  const float* Wv = (const float*)d_in[4];
  const float* Wo = (const float*)d_in[5];
  const float* qw = (const float*)d_in[6];
  const float* kw = (const float*)d_in[7];

  char* ws = (char*)d_ws;
  size_t off = 0;
  auto alloc = [&](size_t bytes) {
    void* p = ws + off;
    off += (bytes + 255) & ~(size_t)255;
    return p;
  };
  bf16*  Xb  = (bf16*)alloc((size_t)TOK * HID * 2);
  bf16*  Wqb = (bf16*)alloc((size_t)HID * HID * 2);
  bf16*  Wkb = (bf16*)alloc((size_t)NKV * HD * HID * 2);
  bf16*  Wvb = (bf16*)alloc((size_t)NKV * HD * HID * 2);
  bf16*  Wob = (bf16*)alloc((size_t)HID * HID * 2);
  float* Qf  = (float*)alloc((size_t)TOK * NH * HD * 4);
  float* Kf  = (float*)alloc((size_t)TOK * NKV * HD * 4);
  bf16*  Qb  = (bf16*)alloc((size_t)TOK * NH * HD * 2);
  bf16*  Kb  = (bf16*)alloc((size_t)TOK * NKV * HD * 2);
  bf16*  Vb  = (bf16*)alloc((size_t)TOK * NKV * HD * 2);
  bf16*  Vt  = (bf16*)alloc((size_t)TOK * NKV * HD * 2);
  bf16*  Ob  = (bf16*)alloc((size_t)TOK * NH * HD * 2);

  auto cvt = [&](const float* in, bf16* out, long n) {
    int n4 = (int)(n / 4);
    int blocks = (n4 + 255) / 256;
    if (blocks > 8192) blocks = 8192;
    cvt_f32_bf16_v4<<<blocks, 256, 0, stream>>>(in, out, n4);
  };
  cvt(hs, Xb, (long)TOK * HID);
  cvt(Wq, Wqb, (long)HID * HID);
  cvt(Wk, Wkb, (long)NKV * HD * HID);
  cvt(Wv, Wvb, (long)NKV * HD * HID);
  cvt(Wo, Wob, (long)HID * HID);

  const dim3 blk(256);
  // QKV projections (V written directly as bf16; Q/K kept fp32 for norm+rope)
  gemm_wmma<false><<<dim3(HID / 256,      TOK / 128), blk, 0, stream>>>(Xb, Wqb, Qf, TOK, HID,      HID);
  gemm_wmma<false><<<dim3(NKV * HD / 256, TOK / 128), blk, 0, stream>>>(Xb, Wkb, Kf, TOK, NKV * HD, HID);
  gemm_wmma<true ><<<dim3(NKV * HD / 256, TOK / 128), blk, 0, stream>>>(Xb, Wvb, Vb, TOK, NKV * HD, HID);

  norm_rope<<<TOK * NH  / 8, 256, 0, stream>>>(Qf, qw, Qb, NH);
  norm_rope<<<TOK * NKV / 8, 256, 0, stream>>>(Kf, kw, Kb, NKV);

  {
    long n = (long)TOK * NKV * HD;
    transpose_v<<<(int)((n + 255) / 256), 256, 0, stream>>>(Vb, Vt);
  }

  flash_attn<<<dim3(SEQ / 128, NH, BATCH), 256, 0, stream>>>(Qb, Kb, Vt, Ob);

  // output projection -> fp32 d_out
  gemm_wmma<false><<<dim3(HID / 256, TOK / 128), blk, 0, stream>>>(Ob, Wob, (float*)d_out, TOK, HID, HID);
}
